// LocalTemporalAttention_11098195493002
// MI455X (gfx1250) — matrix operand order repaired and checked by reading, not compiled, not run on hardware
//
#include <hip/hip_runtime.h>

// ---------------------------------------------------------------------------
// LocalTemporalAttention for MI455X (gfx1250, wave32, WMMA bf16 + TDM)
// B=2, T=64, S=1024, C=512, H=8, hd=64, WINDOW=5
// ---------------------------------------------------------------------------

typedef __attribute__((ext_vector_type(16))) __bf16 v16bf;
typedef __attribute__((ext_vector_type(8)))  float  v8f;
typedef unsigned int v4u __attribute__((ext_vector_type(4)));
typedef int          v4i __attribute__((ext_vector_type(4)));
typedef int          v8i __attribute__((ext_vector_type(8)));

#define BB 2
#define TT 64
#define SS 1024
#define CC 512
#define HH 8
#define HD 64
#define WIN 5

#if __has_builtin(__builtin_amdgcn_tensor_load_to_lds) && \
    __has_builtin(__builtin_amdgcn_tensor_store_from_lds) && \
    __has_builtin(__builtin_amdgcn_s_wait_tensorcnt)
#define USE_TDM 1
#else
#define USE_TDM 0
#endif

union FragU { v16bf v; float4 f4[2]; };

// A-matrix fragment (16x32 bf16, MxK). ISA layout:
// lanes 0-15: M=lane,  K=0..7 in v0-3, K=16..23 in v4-7
// lanes 16-31: M=lane-16, K=8..15 in v0-3, K=24..31 in v4-7
__device__ __forceinline__ v16bf load_frag_a(const __bf16* p, int rowbase, int ld,
                                             int kbase, int lane) {
    int r  = rowbase + (lane & 15);
    int k0 = kbase + ((lane & 16) ? 8 : 0);
    FragU u;
    u.f4[0] = *reinterpret_cast<const float4*>(p + r * ld + k0);
    u.f4[1] = *reinterpret_cast<const float4*>(p + r * ld + k0 + 16);
    return u.v;
}

// B-matrix fragment (32x16 bf16, KxN), loaded from a transposed (NxK) LDS tile.
// lanes 0-15: N=lane, K=0..7,16..23 ; lanes 16-31: N=lane-16, K=8..15,24..31
__device__ __forceinline__ v16bf load_frag_b(const __bf16* p, int rowbase, int ld,
                                             int kbase, int lane) {
    int r  = rowbase + (lane & 15);
    int k0 = kbase + ((lane & 16) ? 8 : 0);
    FragU u;
    u.f4[0] = *reinterpret_cast<const float4*>(p + r * ld + k0);
    u.f4[1] = *reinterpret_cast<const float4*>(p + r * ld + k0 + 16);
    return u.v;
}

#if USE_TDM
// ---------------------------------------------------------------------------
// Build a 2D-tile Tensor-DMA descriptor (D#) per CDNA5 ISA 8.3-8.6 and issue
// TENSOR_LOAD_TO_LDS / TENSOR_STORE_FROM_LDS (6-arg builtin: g0,g1,g2,g3,g4,cpol).
//   tile: rows x width bf16 elems, global row stride = row_stride (elems),
//   LDS destination packed row-major (row stride = width elems).
// ---------------------------------------------------------------------------
__device__ __forceinline__ void tdm_make_desc(const void* gaddr, unsigned lds_off,
                                              unsigned width, unsigned rows,
                                              unsigned long long row_stride,
                                              v4u& g0, v8i& g1) {
    unsigned long long ga = (unsigned long long)gaddr;
    g0.x = 1u;                                          // count=1, user mode
    g0.y = lds_off;                                     // LDS byte address
    g0.z = (unsigned)(ga & 0xffffffffu);                // global_addr[31:0]
    g0.w = (unsigned)((ga >> 32) & 0x1ffffffu) | (2u << 30);  // addr[56:32]|type=2
    g1[0] = (int)(1u << 16);                            // data_size=1 -> 2 bytes
    g1[1] = (int)(width << 16);                         // tensor_dim0[15:0]
    g1[2] = (int)(rows << 16);                          // tensor_dim1[15:0]
    g1[3] = (int)(width << 16);                         // tile_dim0
    g1[4] = (int)rows;                                  // tile_dim1 (tile_dim2=0)
    g1[5] = (int)(unsigned)(row_stride & 0xffffffffu);  // tensor_dim0_stride lo
    g1[6] = (int)(unsigned)((row_stride >> 32) & 0xffffu); // stride hi
    g1[7] = 0;
}

__device__ __forceinline__ void tdm_tile_load(const void* gaddr, unsigned lds_off,
                                              unsigned width, unsigned rows,
                                              unsigned long long row_stride) {
    v4u g0; v8i g1;
    tdm_make_desc(gaddr, lds_off, width, rows, row_stride, g0, g1);
    v4i g2 = {0, 0, 0, 0};
    v4i g3 = {0, 0, 0, 0};
    v8i g4 = {0, 0, 0, 0, 0, 0, 0, 0};
    __builtin_amdgcn_tensor_load_to_lds(g0, g1, g2, g3, g4, 0);
}

__device__ __forceinline__ void tdm_tile_store(const void* gaddr, unsigned lds_off,
                                               unsigned width, unsigned rows,
                                               unsigned long long row_stride) {
    v4u g0; v8i g1;
    tdm_make_desc(gaddr, lds_off, width, rows, row_stride, g0, g1);
    v4i g2 = {0, 0, 0, 0};
    v4i g3 = {0, 0, 0, 0};
    v8i g4 = {0, 0, 0, 0, 0, 0, 0, 0};
    __builtin_amdgcn_tensor_store_from_lds(g0, g1, g2, g3, g4, 0);
}
#endif

// ---------------------------------------------------------------------------
// Tiled GEMM: C[M x N] = A[M x K] * Bw[K x N]
// A: fp32 or bf16 (converted to bf16 in LDS), Bw: fp32, C: bf16 or fp32.
// Block = 256 threads (8 waves), tile 128x64, K-step 32.
// ---------------------------------------------------------------------------
template <typename TA, typename TC>
__global__ __launch_bounds__(256) void gemm_bf16_wmma(
    const TA* __restrict__ A, const float* __restrict__ Bw, TC* __restrict__ C,
    int K, int lda, int ldb, int ldc) {
    __shared__ __bf16 sA[128 * 32];   // row-major MxK tile
    __shared__ __bf16 sBt[64 * 32];   // transposed NxK tile

    const int tid  = threadIdx.x;
    const int lane = tid & 31;
    const int w    = tid >> 5;
    const long mBase = (long)blockIdx.x * 128;
    const int  nBase = blockIdx.y * 64;

    v8f acc[4] = {v8f{}, v8f{}, v8f{}, v8f{}};

    const int nkt = K >> 5;
    for (int kt = 0; kt < nkt; ++kt) {
        // ---- stage A tile (128x32) ----
        {
            const int c4 = (tid & 7) * 4;
            const int rr = tid >> 3;
#pragma unroll
            for (int r0 = 0; r0 < 128; r0 += 32) {
                const int row = r0 + rr;
                const long gidx = (mBase + row) * (long)lda + kt * 32 + c4;
                if constexpr (sizeof(TA) == 4) {
                    float4 vv = *reinterpret_cast<const float4*>(A + gidx);
                    sA[row * 32 + c4 + 0] = (__bf16)vv.x;
                    sA[row * 32 + c4 + 1] = (__bf16)vv.y;
                    sA[row * 32 + c4 + 2] = (__bf16)vv.z;
                    sA[row * 32 + c4 + 3] = (__bf16)vv.w;
                } else {
                    *reinterpret_cast<uint2*>(&sA[row * 32 + c4]) =
                        *reinterpret_cast<const uint2*>(A + gidx);
                }
                if (kt + 1 < nkt)   // prefetch next K-tile (global_prefetch_b8)
                    __builtin_prefetch(A + gidx + 32, 0, 1);
            }
        }
        // ---- stage B tile (32x64 -> transposed 64x32) ----
        {
            const int c4 = (tid & 15) * 4;
            const int kr = tid >> 4;
#pragma unroll
            for (int k0 = 0; k0 < 32; k0 += 16) {
                const int krow = k0 + kr;
                const long gidx = (long)(kt * 32 + krow) * ldb + nBase + c4;
                float4 vv = *reinterpret_cast<const float4*>(Bw + gidx);
                sBt[(c4 + 0) * 32 + krow] = (__bf16)vv.x;
                sBt[(c4 + 1) * 32 + krow] = (__bf16)vv.y;
                sBt[(c4 + 2) * 32 + krow] = (__bf16)vv.z;
                sBt[(c4 + 3) * 32 + krow] = (__bf16)vv.w;
                if (kt + 1 < nkt)
                    __builtin_prefetch(Bw + gidx + 32 * (long)ldb, 0, 1);
            }
        }
        __syncthreads();

        v16bf a = load_frag_a(sA, w * 16, 32, 0, lane);
#pragma unroll
        for (int j = 0; j < 4; ++j) {
            v16bf bf = load_frag_b(sBt, j * 16, 32, 0, lane);
            acc[j] = __builtin_amdgcn_wmma_f32_16x16x32_bf16(
                false, a, false, bf, (short)0, acc[j], false, false);
        }
        __syncthreads();
    }

    // ---- store C ----
    const int rhalf = (lane & 16) ? 8 : 0;
    const int cb    = lane & 15;
#pragma unroll
    for (int j = 0; j < 4; ++j) {
#pragma unroll
        for (int i = 0; i < 8; ++i) {
            const long row = mBase + w * 16 + i + rhalf;
            const int  col = nBase + j * 16 + cb;
            if constexpr (sizeof(TC) == 4)
                C[row * (long)ldc + col] = acc[j][i];
            else
                C[row * (long)ldc + col] = (TC)acc[j][i];
        }
    }
}

// ---------------------------------------------------------------------------
// Local-window attention: one block per (b,h,s). T=64, hd=64, window=5.
// qkv layout: ((b*T+t)*S+s)*1536 + part*512 + h*64 + d   (bf16)
// out layout: ((b*T+t)*S+s)*512 + h*64 + d               (bf16)
// Q/K tiles loaded by the Tensor Data Mover (waves 0/1); V transposed by
// waves 2/3; result staged in LDS and written back by one tensor store.
// ---------------------------------------------------------------------------
__global__ __launch_bounds__(128) void local_attn_wmma(
    const __bf16* __restrict__ qkv, __bf16* __restrict__ out) {
    __shared__ __bf16 sQ[64 * 64];   // t x d
    __shared__ __bf16 sK[64 * 64];   // u x d  (acts as transposed B)
    __shared__ __bf16 sVt[64 * 64];  // d x u  (transposed V, acts as Bt)
    __shared__ __bf16 sP[64 * 64];   // t x u probs, then reused for out tile

    const int bid = blockIdx.x;
    const int s = bid & (SS - 1);
    const int h = (bid >> 10) & (HH - 1);
    const int b = bid >> 13;

    const int tid  = threadIdx.x;
    const int lane = tid & 31;
    const int w    = tid >> 5;

    const long rowStrideQKV = (long)SS * (3 * CC);                 // elems per t
    const long base0 = ((long)(b * TT) * SS + s) * (3 * CC) + h * HD;

#if USE_TDM
    {
        // wave-uniform (SGPR) branch: TDM ignores EXEC, so issue exactly once.
        const int wu = __builtin_amdgcn_readfirstlane(tid) >> 5;
        if (wu == 0) {
            tdm_tile_load(qkv + base0 + 0 * CC,
                          (unsigned)(unsigned long long)(unsigned long)(size_t)&sQ[0],
                          HD, TT, (unsigned long long)rowStrideQKV);
        } else if (wu == 1) {
            tdm_tile_load(qkv + base0 + 1 * CC,
                          (unsigned)(unsigned long long)(unsigned long)(size_t)&sK[0],
                          HD, TT, (unsigned long long)rowStrideQKV);
        } else {
            // waves 2,3: transpose-copy V (64 threads -> one row each)
            const int row = tid - 64;
            const long vb = base0 + (long)row * rowStrideQKV + 2 * CC;
#pragma unroll
            for (int e = 0; e < 8; ++e) {
                const int d0 = e * 8;
                union { uint4 u4; __bf16 hv[8]; } vv;
                vv.u4 = *reinterpret_cast<const uint4*>(qkv + vb + d0);
#pragma unroll
                for (int jj = 0; jj < 8; ++jj)
                    sVt[(d0 + jj) * 64 + row] = vv.hv[jj];
            }
        }
        __builtin_amdgcn_s_wait_tensorcnt(0);
    }
#else
    {
        const int row   = tid >> 1;
        const int dbase = (tid & 1) * 32;
        const long base = base0 + (long)row * rowStrideQKV;
#pragma unroll
        for (int e = 0; e < 4; ++e) {
            const int d0 = dbase + e * 8;
            *reinterpret_cast<uint4*>(&sQ[row * 64 + d0]) =
                *reinterpret_cast<const uint4*>(qkv + base + 0 * CC + d0);
            *reinterpret_cast<uint4*>(&sK[row * 64 + d0]) =
                *reinterpret_cast<const uint4*>(qkv + base + 1 * CC + d0);
            union { uint4 u4; __bf16 hv[8]; } vv;
            vv.u4 = *reinterpret_cast<const uint4*>(qkv + base + 2 * CC + d0);
#pragma unroll
            for (int jj = 0; jj < 8; ++jj)
                sVt[(d0 + jj) * 64 + row] = vv.hv[jj];
        }
    }
#endif
    __syncthreads();

    // ---- scores = Q * K^T  (each wave: rows [w*16, w*16+16) x 64 cols) ----
    v8f acc[4] = {v8f{}, v8f{}, v8f{}, v8f{}};
#pragma unroll
    for (int ks = 0; ks < 64; ks += 32) {
        v16bf a = load_frag_a(sQ, w * 16, 64, ks, lane);
#pragma unroll
        for (int j = 0; j < 4; ++j) {
            v16bf bf = load_frag_b(sK, j * 16, 64, ks, lane);
            acc[j] = __builtin_amdgcn_wmma_f32_16x16x32_bf16(
                false, a, false, bf, (short)0, acc[j], false, false);
        }
    }

    // ---- windowed mask + softmax (row spread over 16 lanes x 4 accs) ----
    const int rhalf = (lane & 16) ? 8 : 0;
    const int cb    = lane & 15;
    const float scale = 0.125f;   // hd^-0.5 = 1/8
#pragma unroll
    for (int i = 0; i < 8; ++i) {
        const int trow = w * 16 + i + rhalf;
        float vals[4];
#pragma unroll
        for (int j = 0; j < 4; ++j) {
            const int u = j * 16 + cb;
            const int d = trow - u;
            vals[j] = (d <= WIN && d >= -WIN) ? acc[j][i] * scale : -3.0e38f;
        }
        float m = fmaxf(fmaxf(vals[0], vals[1]), fmaxf(vals[2], vals[3]));
#pragma unroll
        for (int msk = 1; msk < 16; msk <<= 1)
            m = fmaxf(m, __shfl_xor(m, msk, 32));
        float e[4];
        float sum = 0.0f;
#pragma unroll
        for (int j = 0; j < 4; ++j) { e[j] = __expf(vals[j] - m); sum += e[j]; }
#pragma unroll
        for (int msk = 1; msk < 16; msk <<= 1)
            sum += __shfl_xor(sum, msk, 32);
        const float inv = 1.0f / sum;
#pragma unroll
        for (int j = 0; j < 4; ++j)
            sP[trow * 64 + j * 16 + cb] = (__bf16)(e[j] * inv);
    }
    __syncthreads();

    // ---- out = P * V  (B-tile from transposed V) ----
    v8f acc2[4] = {v8f{}, v8f{}, v8f{}, v8f{}};
#pragma unroll
    for (int ks = 0; ks < 64; ks += 32) {
        v16bf a = load_frag_a(sP, w * 16, 64, ks, lane);
#pragma unroll
        for (int j = 0; j < 4; ++j) {
            v16bf bf = load_frag_b(sVt, j * 16, 64, ks, lane);
            acc2[j] = __builtin_amdgcn_wmma_f32_16x16x32_bf16(
                false, a, false, bf, (short)0, acc2[j], false, false);
        }
    }
    __syncthreads();   // done reading sP as probs; reuse it as output staging

#pragma unroll
    for (int j = 0; j < 4; ++j) {
#pragma unroll
        for (int i = 0; i < 8; ++i) {
            const int trow = w * 16 + i + rhalf;
            const int d    = j * 16 + cb;
#if USE_TDM
            sP[trow * 64 + d] = (__bf16)acc2[j][i];
#else
            out[(((long)(b * TT + trow) * SS + s) * CC) + h * HD + d] =
                (__bf16)acc2[j][i];
#endif
        }
    }

#if USE_TDM
    __syncthreads();
    const int wu2 = __builtin_amdgcn_readfirstlane(tid) >> 5;
    if (wu2 == 0) {
        const long obase = ((long)(b * TT) * SS + s) * CC + h * HD;
        tdm_tile_store(out + obase,
                       (unsigned)(unsigned long long)(unsigned long)(size_t)&sP[0],
                       HD, TT, (unsigned long long)SS * CC);
        __builtin_amdgcn_s_wait_tensorcnt(0);
    }
#endif
}

// ---------------------------------------------------------------------------
extern "C" void kernel_launch(void* const* d_in, const int* in_sizes, int n_in,
                              void* d_out, int out_size, void* d_ws, size_t ws_size,
                              hipStream_t stream) {
    const float* x    = (const float*)d_in[0];   // (2, 65536, 512)
    const float* Wqkv = (const float*)d_in[1];   // (512, 1536)
    const float* Wout = (const float*)d_in[2];   // (512, 512)
    (void)in_sizes; (void)n_in; (void)out_size; (void)ws_size;

    const long M = (long)BB * TT * SS;           // 131072

    __bf16* qkv   = (__bf16*)d_ws;                                   // M*1536 bf16
    __bf16* attno = (__bf16*)((char*)d_ws + (size_t)M * 1536 * 2);   // M*512 bf16
    float*  outp  = (float*)d_out;

    // 1) qkv = x @ W_qkv  (fp32 in, bf16 out)
    gemm_bf16_wmma<float, __bf16>
        <<<dim3(M / 128, (3 * CC) / 64), 256, 0, stream>>>(
            x, Wqkv, qkv, CC, CC, 3 * CC, 3 * CC);

    // 2) windowed attention per (b,h,s)
    local_attn_wmma<<<BB * HH * SS, 128, 0, stream>>>(qkv, attno);

    // 3) out = attn_out @ W_out  (bf16 in, fp32 out)
    gemm_bf16_wmma<__bf16, float>
        <<<dim3(M / 128, CC / 64), 256, 0, stream>>>(
            attno, Wout, outp, CC, CC, CC, CC);
}
